// VllmMixtureOfExpertsOpFP8_15401752723973
// MI455X (gfx1250) — matrix operand order repaired and checked by reading, not compiled, not run on hardware
//
#include <hip/hip_runtime.h>
#include <hip/hip_bf16.h>
#include <math.h>

// ---------------------------------------------------------------------------
// MoE FFN (fp32 weights w/ 128x128 block scales) for gfx1250 (MI455X, wave32)
// routing -> GEMM1 (TDM/async-LDS + bf16 WMMA) -> silu*up -> GEMM2 -> combine
//   - weight tiles  : Tensor Data Mover (tensor_load_to_lds, TENSORcnt)
//   - gathered acts : global_load_async_to_lds_b128 (ASYNCcnt)
//   - math          : v_wmma_f32_16x16x32_bf16, dequant folded into cvt
// ---------------------------------------------------------------------------

typedef __attribute__((ext_vector_type(16))) __bf16 v16bf;
typedef __attribute__((ext_vector_type(8)))  float  v8f;
typedef unsigned int v4u __attribute__((ext_vector_type(4)));
typedef int          v8i __attribute__((ext_vector_type(8)));
typedef int          v4i __attribute__((ext_vector_type(4)));

#define T_TOK   256
#define H_DIM   2048
#define I_DIM   1024
#define N_EXP   8
#define TOPK    2
#define NSLOT   (T_TOK * TOPK)   // 512

#define BM    64
#define BN    128
#define BK    32
#define LDKF  (BK + 4)           // fp32 LDS row stride (36 floats = 144 B); TDM pad reproduces this
#define LDC_S (BN + 4)           // epilogue LDS row stride (132 floats)

// ---------------------------------------------------------------------------
// Routing: deterministic compaction of the 512 (token,k) slots by expert.
// ---------------------------------------------------------------------------
__global__ __launch_bounds__(NSLOT)
void moe_routing(const int* __restrict__ topk_ids,
                 int* __restrict__ counts, int* __restrict__ base,
                 int* __restrict__ tok, int* __restrict__ slot_of)
{
    __shared__ int ids[NSLOT];
    const int i = threadIdx.x;
    ids[i] = topk_ids[i];
    if (i < N_EXP) { counts[i] = 0; base[i] = 0; }
    __syncthreads();

    const int e = ids[i];
    int rank = 0, less = 0, cnt = 0;
    for (int j = 0; j < NSLOT; ++j) {
        const int ej = ids[j];
        rank += (int)((j < i) & (ej == e));
        less += (int)(ej < e);
        cnt  += (int)(ej == e);
    }
    const int cidx = less + rank;
    tok[cidx]  = i >> 1;          // token id (TOPK = 2)
    slot_of[i] = cidx;
    if (rank == 0) { base[e] = less; counts[e] = cnt; }
}

// ---------------------------------------------------------------------------
// Async global -> LDS copy (16 bytes per lane), tracked with ASYNCcnt.
// ---------------------------------------------------------------------------
__device__ __forceinline__ void async_b128(uint32_t lds_off, const float* g)
{
    asm volatile("global_load_async_to_lds_b128 %0, %1, off"
                 :: "v"(lds_off), "v"((unsigned long long)(uintptr_t)g)
                 : "memory");
}

__device__ __forceinline__ uint32_t lds_off32(const void* p)
{
    // generic pointer to a __shared__ object: low 32 bits = LDS byte offset
    return (uint32_t)(uintptr_t)p;
}

// ---------------------------------------------------------------------------
// TDM: 2-D tile load (rows x BK fp32, global row stride = ktot elements) into
// LDS with hardware row padding (pad every 32 DWORDs by 4 DWORDs -> LDKF=36).
// D# layout per CDNA5 ISA 8.3-8.6. Issued by one wave; tracked by TENSORcnt.
// ---------------------------------------------------------------------------
__device__ __forceinline__ void tdm_load_tile(uint32_t lds_byte_off,
                                              unsigned long long gaddr,
                                              int ktot, int nout, int rows)
{
    v4u g0;
    g0[0] = 1u;                                            // count=1, user mode
    g0[1] = lds_byte_off;                                  // lds_addr (bytes)
    g0[2] = (uint32_t)gaddr;                               // global_addr[31:0]
    g0[3] = (uint32_t)((gaddr >> 32) & 0x01FFFFFFu)        // global_addr[56:32]
          | (2u << 30);                                    // type = 2 ("image")
    v8i g1;
    g1[0] = (int)((2u << 16)        // data_size = 2 -> 4 bytes
                | (1u << 20)        // pad_enable
                | (4u << 22)        // pad_interval: 32 DWORDs (one 128B row)
                | (3u << 25));      // pad_amount: 4 DWORDs (16 B) -> stride 144B
    g1[1] = (int)(((uint32_t)ktot & 0xFFFFu) << 16);                 // tensor_dim0[15:0]
    g1[2] = (int)(((uint32_t)ktot >> 16) | (((uint32_t)nout) << 16));// dim0[31:16] | dim1[15:0]
    g1[3] = (int)(((uint32_t)BK) << 16);                             // tile_dim0 = 32
    g1[4] = rows;                                                    // tile_dim1
    g1[5] = ktot;                                                    // dim0_stride[31:0]
    g1[6] = 0;                                                       // stride0[47:32]|stride1 lo
    g1[7] = 0;
    v4i g2 = {0, 0, 0, 0};
    v4i g3 = {0, 0, 0, 0};
#if __clang_major__ >= 23
    v8i gz = {0, 0, 0, 0, 0, 0, 0, 0};
    __builtin_amdgcn_tensor_load_to_lds(g0, g1, g2, g3, gz, 0);
#else
    __builtin_amdgcn_tensor_load_to_lds(g0, g1, g2, g3, 0);
#endif
}

// ---------------------------------------------------------------------------
// WMMA fragment builders from fp32 LDS tiles (convert + dequant at read time).
// A (16x32): lane L -> row L%16, K = {h*16 .. h*16+15}, h = L/16.
// B (32x16): lane L -> weight row L%16, K = {h*16 .. h*16+15}.
// ---------------------------------------------------------------------------
__device__ __forceinline__ v16bf frag_a(const float* rows, int lane)
{
    const int r = lane & 15, h = lane >> 4;
    const float* p = rows + r * LDKF;
    const float4 q0 = *(const float4*)(p + h * 16);
    const float4 q1 = *(const float4*)(p + h * 16 + 4);
    const float4 q2 = *(const float4*)(p + h * 16 + 8);
    const float4 q3 = *(const float4*)(p + h * 16 + 12);
    v16bf v;
    v[ 0]=(__bf16)q0.x; v[ 1]=(__bf16)q0.y; v[ 2]=(__bf16)q0.z; v[ 3]=(__bf16)q0.w;
    v[ 4]=(__bf16)q1.x; v[ 5]=(__bf16)q1.y; v[ 6]=(__bf16)q1.z; v[ 7]=(__bf16)q1.w;
    v[ 8]=(__bf16)q2.x; v[ 9]=(__bf16)q2.y; v[10]=(__bf16)q2.z; v[11]=(__bf16)q2.w;
    v[12]=(__bf16)q3.x; v[13]=(__bf16)q3.y; v[14]=(__bf16)q3.z; v[15]=(__bf16)q3.w;
    return v;
}

__device__ __forceinline__ v16bf frag_b(const float* rows, int lane, float s)
{
    const int r = lane & 15, h = lane >> 4;
    const float* p = rows + r * LDKF + h * 16;
    const float4 q0 = *(const float4*)(p);
    const float4 q1 = *(const float4*)(p + 4);
    const float4 q2 = *(const float4*)(p + 8);
    const float4 q3 = *(const float4*)(p + 12);
    v16bf v;
    v[ 0]=(__bf16)(q0.x*s); v[ 1]=(__bf16)(q0.y*s); v[ 2]=(__bf16)(q0.z*s); v[ 3]=(__bf16)(q0.w*s);
    v[ 4]=(__bf16)(q1.x*s); v[ 5]=(__bf16)(q1.y*s); v[ 6]=(__bf16)(q1.z*s); v[ 7]=(__bf16)(q1.w*s);
    v[ 8]=(__bf16)(q2.x*s); v[ 9]=(__bf16)(q2.y*s); v[10]=(__bf16)(q2.z*s); v[11]=(__bf16)(q2.w*s);
    v[12]=(__bf16)(q3.x*s); v[13]=(__bf16)(q3.y*s); v[14]=(__bf16)(q3.z*s); v[15]=(__bf16)(q3.w*s);
    return v;
}

// ---------------------------------------------------------------------------
// Per-expert tiled GEMM:  C[r, n] = sum_k A[sel(r), k] * (Bw[e, n, k] * scale)
// Block 64x128, K-chunk 32, double-buffered (TDM for B, async-LDS for A),
// 8 waves (2x4), each wave 32x32 (2x2 WMMA). Coalesced LDS-staged epilogue.
// ---------------------------------------------------------------------------
struct SmemT {
    union {
        struct { float A[2][BM * LDKF]; float B[2][BN * LDKF]; } s;  // 55296 B
        float c[BM * LDC_S];                                         // 33792 B
    };
    int toks[BM];
};

template<bool GATHER, int KTOT, int LDA, int NOUT>
__global__ __launch_bounds__(256)
void moe_gemm(const float* __restrict__ A,
              const float* __restrict__ Bw,
              const float* __restrict__ Bscale,
              float* __restrict__ C,
              const int* __restrict__ counts,
              const int* __restrict__ base,
              const int* __restrict__ tok)
{
    __shared__ SmemT sm;

    const int e   = blockIdx.z;
    const int cnt = counts[e];
    const int r0  = blockIdx.y * BM;
    if (r0 >= cnt) return;                 // early-exit: static grid, dynamic M
    const int cb  = base[e];
    const int n0  = blockIdx.x * BN;

    const int tid  = threadIdx.x;
    const int lane = tid & 31;
    const int w    = tid >> 5;
    const int wm   = w & 1;                // 2 wave-rows  -> BM=64
    const int wn   = w >> 1;               // 4 wave-cols  -> BN=128
    const bool is_w0 = (w == 0);

    if (tid < BM) {
        const int r = r0 + tid;
        int cidx = cb + r;
        if (cidx > NSLOT - 1) cidx = NSLOT - 1;
        sm.toks[tid] = GATHER ? ((r < cnt) ? tok[cidx] : 0) : cidx;
    }
    __syncthreads();

    const float* Be   = Bw     + (size_t)e * NOUT * KTOT;
    const float* Se   = Bscale + (size_t)e * (NOUT / 128) * (KTOT / 128);
    const int    nblk = blockIdx.x;        // BN==128 -> n-scale-block id

    // A: per-thread async endpoints (2 x b128 per tile)
    const float* aptr[2]; uint32_t aoff[2];
    #pragma unroll
    for (int it = 0; it < 2; ++it) {
        const int q = tid + it * 256, row = q >> 3, c4 = (q & 7) * 4;
        aptr[it] = A + (size_t)sm.toks[row] * LDA + c4;
        aoff[it] = lds_off32(&sm.s.A[0][row * LDKF + c4]);
    }
    constexpr uint32_t ABUF = (uint32_t)(BM * LDKF * 4);   // byte stride buf0->buf1

    // B: one TDM descriptor per tile (wave 0 only)
    const unsigned long long bga = (unsigned long long)(uintptr_t)(Be + (size_t)n0 * KTOT);
    const uint32_t blds0 = lds_off32(&sm.s.B[0][0]);
    const uint32_t blds1 = lds_off32(&sm.s.B[1][0]);

    auto issue_a = [&](int buf, int k0) {
        #pragma unroll
        for (int it = 0; it < 2; ++it)
            async_b128(aoff[it] + (uint32_t)buf * ABUF, aptr[it] + k0);
    };
    auto issue_b = [&](int buf, int k0) {
        tdm_load_tile(buf ? blds1 : blds0, bga + (unsigned long long)k0 * 4u,
                      KTOT, NOUT, BN);
    };

    v8f acc[2][2] = {};

    auto compute = [&](int buf, int k0) {
        const float sB = Se[nblk * (KTOT / 128) + (k0 >> 7)];
        const v16bf a0 = frag_a(&sm.s.A[buf][(wm * 32 +  0) * LDKF], lane);
        const v16bf a1 = frag_a(&sm.s.A[buf][(wm * 32 + 16) * LDKF], lane);
        const v16bf b0 = frag_b(&sm.s.B[buf][(wn * 32 +  0) * LDKF], lane, sB);
        const v16bf b1 = frag_b(&sm.s.B[buf][(wn * 32 + 16) * LDKF], lane, sB);
        acc[0][0] = __builtin_amdgcn_wmma_f32_16x16x32_bf16(false, a0, false, b0, (short)0, acc[0][0], false, false);
        acc[0][1] = __builtin_amdgcn_wmma_f32_16x16x32_bf16(false, a0, false, b1, (short)0, acc[0][1], false, false);
        acc[1][0] = __builtin_amdgcn_wmma_f32_16x16x32_bf16(false, a1, false, b0, (short)0, acc[1][0], false, false);
        acc[1][1] = __builtin_amdgcn_wmma_f32_16x16x32_bf16(false, a1, false, b1, (short)0, acc[1][1], false, false);
    };

    constexpr int NT = KTOT / BK;
    int buf = 0;
    issue_a(0, 0);
    if (is_w0) issue_b(0, 0);

    for (int i = 0; i < NT - 1; ++i) {                      // steady state
        const int k0 = i * BK;
        issue_a(buf ^ 1, k0 + BK);                          // prefetch next tile
        if (is_w0) issue_b(buf ^ 1, k0 + BK);
        asm volatile("s_wait_asynccnt 2" ::: "memory");     // tile i's A landed
        if (is_w0) __builtin_amdgcn_s_wait_tensorcnt(1);    // tile i's B landed
        __syncthreads();                                    // cross-wave visibility
        compute(buf, k0);
        __syncthreads();                                    // protect buf^1 overwrite
        buf ^= 1;
    }
    asm volatile("s_wait_asynccnt 0" ::: "memory");         // peeled last tile
    if (is_w0) __builtin_amdgcn_s_wait_tensorcnt(0);
    __syncthreads();
    compute(buf, (NT - 1) * BK);
    __syncthreads();                                        // before LDS reuse

    // --- epilogue: accumulators -> LDS (padded) -> coalesced b128 stores ----
    // D layout (16x16 f32): lane L, VGPR v -> M = (L>>4)*8 + v, N = L&15
    {
        const int mb = wm * 32 + ((lane >> 4) * 8);
        const int nb = wn * 32 + (lane & 15);
        #pragma unroll
        for (int i = 0; i < 2; ++i)
            #pragma unroll
            for (int j = 0; j < 2; ++j)
                #pragma unroll
                for (int v = 0; v < 8; ++v)
                    sm.c[(mb + i * 16 + v) * LDC_S + nb + j * 16] = acc[i][j][v];
    }
    __syncthreads();

    int act_rows = cnt - r0;
    if (act_rows > BM) act_rows = BM;
    #pragma unroll
    for (int it = 0; it < 8; ++it) {                        // 64 rows x 32 float4
        const int q   = tid + it * 256;
        const int row = q >> 5;
        const int c4  = (q & 31) * 4;
        if (row < act_rows) {
            const float4 val = *(const float4*)(&sm.c[row * LDC_S + c4]);
            *(float4*)(&C[(size_t)(cb + r0 + row) * NOUT + n0 + c4]) = val;
        }
    }
}

// ---------------------------------------------------------------------------
// act[s, i] = silu(gate_up[s, i]) * gate_up[s, 1024 + i]
// ---------------------------------------------------------------------------
__global__ __launch_bounds__(256)
void moe_act(const float* __restrict__ gu, float* __restrict__ act)
{
    const int idx = blockIdx.x * 256 + threadIdx.x;    // NSLOT * I_DIM
    const int s = idx >> 10;
    const int i = idx & (I_DIM - 1);
    const float g = gu[(size_t)s * (2 * I_DIM) + i];
    const float u = gu[(size_t)s * (2 * I_DIM) + I_DIM + i];
    const float sig = 1.0f / (1.0f + __expf(-g));
    act[idx] = g * sig * u;
}

// ---------------------------------------------------------------------------
// out[t, h] = sum_k topk_w[t,k] * y[slot_of[t*2+k], h]
// ---------------------------------------------------------------------------
__global__ __launch_bounds__(256)
void moe_combine(const float* __restrict__ y,
                 const float* __restrict__ tw,
                 const int* __restrict__ slot_of,
                 float* __restrict__ out)
{
    const int idx = blockIdx.x * 256 + threadIdx.x;    // T_TOK * H_DIM
    const int t = idx >> 11;
    const int h = idx & (H_DIM - 1);
    const int s0 = slot_of[t * 2 + 0];
    const int s1 = slot_of[t * 2 + 1];
    out[idx] = tw[t * 2 + 0] * y[(size_t)s0 * H_DIM + h]
             + tw[t * 2 + 1] * y[(size_t)s1 * H_DIM + h];
}

// ---------------------------------------------------------------------------
extern "C" void kernel_launch(void* const* d_in, const int* in_sizes, int n_in,
                              void* d_out, int out_size, void* d_ws, size_t ws_size,
                              hipStream_t stream)
{
    (void)in_sizes; (void)n_in; (void)out_size; (void)ws_size;

    const float* x       = (const float*)d_in[0];
    const int*   topk_id = (const int*)  d_in[1];
    const float* topk_w  = (const float*)d_in[2];
    const float* w13     = (const float*)d_in[3];
    const float* w13_s   = (const float*)d_in[4];
    const float* w2      = (const float*)d_in[5];
    const float* w2_s    = (const float*)d_in[6];
    float*       out     = (float*)d_out;

    char* ws = (char*)d_ws;
    int*  counts  = (int*)ws;            // 8
    int*  base    = counts + 8;          // 8
    int*  tok     = base + 8;            // 512
    int*  slot_of = tok + NSLOT;         // 512
    float* gate_up = (float*)(ws + 4096);                         // 512*2048 f32 = 4 MB
    float* act_ws  = gate_up + (size_t)NSLOT * 2 * I_DIM;         // 512*1024 f32 = 2 MB
    float* y_ws    = act_ws  + (size_t)NSLOT * I_DIM;             // 512*2048 f32 = 4 MB

    moe_routing<<<1, NSLOT, 0, stream>>>(topk_id, counts, base, tok, slot_of);

    // GEMM1: gate_up[cidx, 0..2048) = x[tok] . w13[e]^T   (K = H = 2048)
    dim3 g1(2 * I_DIM / BN, NSLOT / BM, N_EXP);
    moe_gemm<true, H_DIM, H_DIM, 2 * I_DIM><<<g1, 256, 0, stream>>>(
        x, w13, w13_s, gate_up, counts, base, tok);

    moe_act<<<NSLOT * I_DIM / 256, 256, 0, stream>>>(gate_up, act_ws);

    // GEMM2: y[cidx, 0..2048) = act . w2[e]^T   (K = I = 1024)
    dim3 g2(H_DIM / BN, NSLOT / BM, N_EXP);
    moe_gemm<false, I_DIM, I_DIM, H_DIM><<<g2, 256, 0, stream>>>(
        act_ws, w2, w2_s, y_ws, counts, base, tok);

    moe_combine<<<T_TOK * H_DIM / 256, 256, 0, stream>>>(y_ws, topk_w, slot_of, out);
}
